// cDARA_16836271801069
// MI455X (gfx1250) — matrix operand order repaired and checked by reading, not compiled, not run on hardware
//
#include <hip/hip_runtime.h>
#include <hip/hip_bf16.h>
#include <math.h>

// ---------------------------------------------------------------------------
// Types & helpers
// ---------------------------------------------------------------------------
typedef __attribute__((ext_vector_type(16))) __bf16 v16bf;
typedef __attribute__((ext_vector_type(8)))  __bf16 v8bf;
typedef __attribute__((ext_vector_type(8)))  float  v8f;
typedef __attribute__((ext_vector_type(4)))  int    v4i;

// Builtin expects 'int4 __device__*' (global AS) and the LDS counterpart.
typedef __attribute__((address_space(1))) v4i* gv4i_p;   // global int4*
typedef __attribute__((address_space(3))) v4i* lv4i_p;   // LDS int4*

#define B_     8
#define C_     384
#define HEADS_ 8
#define HD_    48
#define N_     16384      // 128*128
#define CQKV_  1152       // 3*C
#define HW_    128

#if __has_builtin(__builtin_amdgcn_global_load_async_to_lds_b128)
#define HAS_ASYNC_LDS 1
#else
#define HAS_ASYNC_LDS 0
#endif

// 16-byte global -> LDS copy: async (ASYNCcnt, bypasses VGPRs) when available.
__device__ inline void cp16_g2l(const void* g, void* l) {
#if HAS_ASYNC_LDS
  __builtin_amdgcn_global_load_async_to_lds_b128((gv4i_p)g, (lv4i_p)l, 0, 0);
#else
  *reinterpret_cast<uint4*>(l) = *reinterpret_cast<const uint4*>(g);
#endif
}
__device__ inline void async_wait0() {
#if __has_builtin(__builtin_amdgcn_s_wait_asynccnt)
  __builtin_amdgcn_s_wait_asynccnt(0);
#elif HAS_ASYNC_LDS
  asm volatile("s_wait_asynccnt 0x0" ::: "memory");
#endif
}

__device__ inline __bf16 f2bf(float f) {
  union { float f; unsigned u; } in; in.f = f;
  unsigned r = in.u + 0x7fffu + ((in.u >> 16) & 1u);   // RNE
  union { unsigned short u; __bf16 b; } out; out.u = (unsigned short)(r >> 16);
  return out.b;
}
__device__ inline float bf2f(__bf16 b) {
  union { __bf16 b; unsigned short u; } in; in.b = b;
  union { unsigned u; float f; } out; out.u = ((unsigned)in.u) << 16;
  return out.f;
}
__device__ inline v8f zero8() {
  v8f z;
#pragma unroll
  for (int i = 0; i < 8; ++i) z[i] = 0.f;
  return z;
}
__device__ inline v8bf ld8(const __bf16* p) { return *reinterpret_cast<const v8bf*>(p); }
__device__ inline v16bf cat8(v8bf lo, v8bf hi) {
  return __builtin_shufflevector(lo, hi, 0,1,2,3,4,5,6,7,8,9,10,11,12,13,14,15);
}

// ---------------------------------------------------------------------------
// K0: pack/convert pointwise QKV weights+bias to bf16, zero the norm buffer
// ---------------------------------------------------------------------------
__global__ void prep_kernel(const float* __restrict__ wq, const float* __restrict__ wk,
                            const float* __restrict__ wv, const float* __restrict__ bq,
                            const float* __restrict__ bk, const float* __restrict__ bv,
                            __bf16* __restrict__ Wqkv, float* __restrict__ biasqkv,
                            float* __restrict__ norms) {
  int idx = blockIdx.x * blockDim.x + threadIdx.x;
  const int NW = CQKV_ * C_;                 // 442368
  if (idx < NW) {
    int grp = idx / (C_ * C_);
    int rem = idx - grp * (C_ * C_);
    const float* src = (grp == 0) ? wq : (grp == 1) ? wk : wv;
    Wqkv[idx] = f2bf(src[rem]);
  }
  if (idx < CQKV_) {
    const float* sb = (idx < C_) ? bq : (idx < 2 * C_) ? bk : bv;
    biasqkv[idx] = sb[idx % C_];
  }
  if (idx < B_ * 2 * C_) norms[idx] = 0.f;
}

// ---------------------------------------------------------------------------
// K1: dcp[b][384] = gelu(d_c @ w1 + b1) @ w2 + b2   (tiny MLP, one block)
// ---------------------------------------------------------------------------
__global__ void dcp_kernel(const float* __restrict__ d_c, const float* __restrict__ w1,
                           const float* __restrict__ b1, const float* __restrict__ w2,
                           const float* __restrict__ b2, float* __restrict__ dcp) {
  __shared__ float hid[B_ * HD_];            // 384
  int tid = threadIdx.x;                     // 384 threads
  {
    int b = tid / HD_, j = tid - b * HD_;
    float s = b1[j];
#pragma unroll
    for (int c = 0; c < 3; ++c) s += d_c[b * 3 + c] * w1[c * HD_ + j];
    float x3 = s * s * s;
    hid[tid] = 0.5f * s * (1.f + tanhf(0.7978845608f * (s + 0.044715f * x3)));
  }
  __syncthreads();
  int o = tid;                               // 0..383
  for (int b = 0; b < B_; ++b) {
    float s = b2[o];
#pragma unroll 8
    for (int j = 0; j < HD_; ++j) s += hid[b * HD_ + j] * w2[j * C_ + o];
    dcp[b * C_ + o] = s;
  }
}

// ---------------------------------------------------------------------------
// WMMA GEMM:  Y[b][m][n] = sum_k W[b][m][k] * X[b][k][n] + bias[m]
//   W bf16 row-major (MxK), X fp32 or bf16 (KxN, n contiguous), Y fp32 or bf16.
//   Block tile 128x128, K-step 32, 8 waves (2x4 frags each).
//   Double-buffered LDS; A tile streams via global_load_async_to_lds (ASYNCcnt),
//   B tile is converted/transposed in-flight (sync stores).
//   Protocol per iter: s_wait_asynccnt 0 (own wave) -> barrier (all waves have
//   waited, so all LDS writes visible) -> issue next stage -> compute current.
// ---------------------------------------------------------------------------
template <typename XT, typename YT>
__global__ __launch_bounds__(256) void gemm_wmma(
    const __bf16* __restrict__ W, long long wStride,
    const XT* __restrict__ X, long long xStride,
    const float* __restrict__ bias,
    YT* __restrict__ Y, long long yStride,
    int K, int Nn) {
  const int nTile = blockIdx.x * 128;
  const int mTile = blockIdx.y * 128;
  const int b     = blockIdx.z;
  const __bf16* Wb = W + (long long)b * wStride;
  const XT*     Xb = X + (long long)b * xStride;
  YT*           Yb = Y + (long long)b * yStride;

  __shared__ __align__(16) __bf16 As[2][128][40];   // M x K32, padded
  __shared__ __align__(16) __bf16 Bs[2][128][40];   // transposed: n x K32, padded

  const int tid  = threadIdx.x;
  const int wave = tid >> 5;
  const int lane = tid & 31;
  const int half = lane >> 4;
  const int l15  = lane & 15;
  const int waveM = (wave & 3) * 32;   // 4 wave-rows of 32
  const int waveN = (wave >> 2) * 64;  // 2 wave-cols of 64

  v8f acc[2][4];
#pragma unroll
  for (int i = 0; i < 2; ++i)
#pragma unroll
    for (int j = 0; j < 4; ++j) acc[i][j] = zero8();

  auto stageA = [&](int buf, int kk) {   // 128x32 bf16 : 512 x 16B, 2 per thread
#pragma unroll
    for (int i = 0; i < 2; ++i) {
      int chunk = tid + i * 256;
      int row   = chunk >> 2;
      int k8    = (chunk & 3) * 8;
      cp16_g2l(Wb + (long long)(mTile + row) * K + kk + k8, &As[buf][row][k8]);
    }
  };
  auto stageB = [&](int buf, int kk) {   // Bs[n][c] = bf16(X[kk+c][nTile+n])
#pragma unroll
    for (int i = 0; i < 4; ++i) {
      int idx = tid + i * 256;           // 0..1023
      int c   = idx >> 5;                // 0..31
      int n4  = (idx & 31) * 4;          // 0..124
      const XT* xsrc = Xb + (long long)(kk + c) * Nn + nTile + n4;
      if constexpr (sizeof(XT) == 4) {
        float4 xv = *reinterpret_cast<const float4*>(xsrc);
        Bs[buf][n4 + 0][c] = f2bf(xv.x);
        Bs[buf][n4 + 1][c] = f2bf(xv.y);
        Bs[buf][n4 + 2][c] = f2bf(xv.z);
        Bs[buf][n4 + 3][c] = f2bf(xv.w);
        if (kk + 32 < K) __builtin_prefetch(xsrc + 32LL * Nn, 0, 3);
      } else {
#pragma unroll
        for (int j = 0; j < 4; ++j) Bs[buf][n4 + j][c] = ((const __bf16*)xsrc)[j];
      }
    }
  };

  const int NK = K / 32;
  stageA(0, 0);
  stageB(0, 0);

  for (int it = 0; it < NK; ++it) {
    async_wait0();        // own async A-tile writes done
    __syncthreads();      // everyone waited -> tiles visible; prev compute done
    if (it + 1 < NK) {    // stream next tiles while computing current
      stageA((it + 1) & 1, (it + 1) * 32);
      stageB((it + 1) & 1, (it + 1) * 32);
    }
    const int buf = it & 1;

    v16bf aF[2], bF[4];
#pragma unroll
    for (int mi = 0; mi < 2; ++mi) {
      int r = waveM + mi * 16 + l15;
      aF[mi] = cat8(ld8(&As[buf][r][half * 8]), ld8(&As[buf][r][16 + half * 8]));
    }
#pragma unroll
    for (int ni = 0; ni < 4; ++ni) {
      int col = waveN + ni * 16 + l15;
      bF[ni] = cat8(ld8(&Bs[buf][col][half * 8]), ld8(&Bs[buf][col][16 + half * 8]));
    }
#pragma unroll
    for (int mi = 0; mi < 2; ++mi)
#pragma unroll
      for (int ni = 0; ni < 4; ++ni)
        acc[mi][ni] = __builtin_amdgcn_wmma_f32_16x16x32_bf16(
            false, aF[mi], false, bF[ni], (short)0, acc[mi][ni], false, false);
  }

  // ---- epilogue: bias + store ----
#pragma unroll
  for (int mi = 0; mi < 2; ++mi)
#pragma unroll
    for (int ni = 0; ni < 4; ++ni)
#pragma unroll
      for (int e = 0; e < 8; ++e) {
        int row = mTile + waveM + mi * 16 + half * 8 + e;
        int col = nTile + waveN + ni * 16 + l15;
        float v = acc[mi][ni][e] + bias[row];
        if constexpr (sizeof(YT) == 4) Yb[(long long)row * Nn + col] = v;
        else                           Yb[(long long)row * Nn + col] = f2bf(v);
      }
}

// ---------------------------------------------------------------------------
// K3: depthwise 3x3 (SAME, zero pad) on bf16 planes + per-channel sum-of-squares
//     for q (grp 0) and k (grp 1).  One block per (channel, batch) plane.
//     Plane staged to LDS via async copies.
// ---------------------------------------------------------------------------
__global__ __launch_bounds__(256) void dwconv_kernel(
    const __bf16* __restrict__ qkvpre,
    const float* __restrict__ wqd, const float* __restrict__ wkd, const float* __restrict__ wvd,
    const float* __restrict__ bqd, const float* __restrict__ bkd, const float* __restrict__ bvd,
    __bf16* __restrict__ qkv, float* __restrict__ norms) {
  const int ch = blockIdx.x;               // 0..1151
  const int b  = blockIdx.y;               // 0..7
  const int grp = ch / C_;                 // 0=q,1=k,2=v
  const int c   = ch - grp * C_;
  const int tid = threadIdx.x;

  __shared__ __align__(16) __bf16 sm[HW_][136];

  const __bf16* src = qkvpre + ((long long)b * CQKV_ + ch) * N_;
  __bf16*       dst = qkv    + ((long long)b * CQKV_ + ch) * N_;

  // stage plane: 2048 x 16B chunks, 8 per thread (async -> LDS)
#pragma unroll
  for (int i = 0; i < 8; ++i) {
    int idx = tid + i * 256;
    int row = idx >> 4;
    int col8 = (idx & 15) * 8;
    cp16_g2l(src + row * HW_ + col8, &sm[row][col8]);
  }
  async_wait0();
  __syncthreads();

  const float* wsel = (grp == 0) ? wqd : (grp == 1) ? wkd : wvd;
  const float* bsel = (grp == 0) ? bqd : (grp == 1) ? bkd : bvd;
  float w[9];
#pragma unroll
  for (int j = 0; j < 9; ++j) w[j] = wsel[c * 9 + j];
  const float bia = bsel[c];

  float ss = 0.f;
#pragma unroll 4
  for (int i = 0; i < 64; ++i) {
    int p = tid + i * 256;
    int y = p >> 7, x = p & 127;
    float a = bia;
#pragma unroll
    for (int dy = 0; dy < 3; ++dy) {
      int yy = y + dy - 1;
      if (yy < 0 || yy >= HW_) continue;
#pragma unroll
      for (int dx = 0; dx < 3; ++dx) {
        int xx = x + dx - 1;
        if (xx < 0 || xx >= HW_) continue;
        a += w[dy * 3 + dx] * bf2f(sm[yy][xx]);
      }
    }
    dst[p] = f2bf(a);
    ss += a * a;
  }
  if (grp < 2) {
    for (int off = 16; off > 0; off >>= 1) ss += __shfl_down(ss, off);
    if ((tid & 31) == 0)
      atomicAdd(&norms[((long long)b * 2 + grp) * C_ + c], ss);
  }
}

// ---------------------------------------------------------------------------
// K4: per (b,h): S = Q_h K_h^T over N=16384 via WMMA, fold norms/temp/dcp/d_l,
//     softmax over d, then W_eff[b][o][h*48+d] = sum_c prj_w[o][h*48+c]*S[c][d].
// ---------------------------------------------------------------------------
__global__ __launch_bounds__(128) void attn_kernel(
    const __bf16* __restrict__ qkv, const float* __restrict__ norms,
    const float* __restrict__ dcp, const float* __restrict__ d_l,
    const float* __restrict__ temperature, const float* __restrict__ prj_w,
    __bf16* __restrict__ W_eff) {
  const int h = blockIdx.x, b = blockIdx.y;
  const int tid = threadIdx.x;
  const int wave = tid >> 5, lane = tid & 31;
  const int half = lane >> 4, l15 = lane & 15;

  const __bf16* Q  = qkv + ((long long)b * CQKV_ + h * HD_) * N_;
  const __bf16* Kp = qkv + ((long long)b * CQKV_ + C_ + h * HD_) * N_;

  __shared__ float S[HD_][HD_ + 1];
  __shared__ float invk[HD_], rowscale[HD_];
  for (int i = tid; i < HD_ * (HD_ + 1); i += 128) (&S[0][0])[i] = 0.f;

  v8f acc[3][3];
#pragma unroll
  for (int i = 0; i < 3; ++i)
#pragma unroll
    for (int j = 0; j < 3; ++j) acc[i][j] = zero8();

  const int n0 = wave * (N_ / 4);
  for (int n = n0; n < n0 + N_ / 4; n += 32) {
    v16bf aF[3], bF[3];
#pragma unroll
    for (int t = 0; t < 3; ++t) {
      const __bf16* qr = Q + (long long)(t * 16 + l15) * N_ + n;
      aF[t] = cat8(ld8(qr + half * 8), ld8(qr + 16 + half * 8));
      const __bf16* kr = Kp + (long long)(t * 16 + l15) * N_ + n;
      bF[t] = cat8(ld8(kr + half * 8), ld8(kr + 16 + half * 8));
    }
#pragma unroll
    for (int mi = 0; mi < 3; ++mi)
#pragma unroll
      for (int ni = 0; ni < 3; ++ni)
        acc[mi][ni] = __builtin_amdgcn_wmma_f32_16x16x32_bf16(
            false, aF[mi], false, bF[ni], (short)0, acc[mi][ni], false, false);
  }
  __syncthreads();
  // cross-wave reduce into LDS
#pragma unroll
  for (int mi = 0; mi < 3; ++mi)
#pragma unroll
    for (int ni = 0; ni < 3; ++ni)
#pragma unroll
      for (int e = 0; e < 8; ++e)
        atomicAdd(&S[mi * 16 + half * 8 + e][ni * 16 + l15], acc[mi][ni][e]);

  if (tid < HD_) {
    float sk = norms[((long long)b * 2 + 1) * C_ + h * HD_ + tid];
    invk[tid] = 1.f / fmaxf(sqrtf(sk), 1e-12f);
    float sq = norms[((long long)b * 2 + 0) * C_ + h * HD_ + tid];
    rowscale[tid] = (1.f / fmaxf(sqrtf(sq), 1e-12f)) * temperature[h] *
                    dcp[b * C_ + h * HD_ + tid] * d_l[b];
  }
  __syncthreads();

  if (tid < HD_) {                      // softmax over d for row c = tid
    const int c = tid;
    float m = -3.4e38f;
    for (int d = 0; d < HD_; ++d) {
      float v = S[c][d] * rowscale[c] * invk[d];
      S[c][d] = v;
      m = fmaxf(m, v);
    }
    float sum = 0.f;
    for (int d = 0; d < HD_; ++d) { float e = expf(S[c][d] - m); S[c][d] = e; sum += e; }
    float inv = 1.f / sum;
    for (int d = 0; d < HD_; ++d) S[c][d] *= inv;
  }
  __syncthreads();

  // W_eff[b][o][h*48+d] = sum_c prj_w[o][h*48+c] * S[c][d]
  for (int i = 0; i < 3; ++i) {
    int o = tid + i * 128;
    float pw[HD_];
#pragma unroll 8
    for (int c = 0; c < HD_; ++c) pw[c] = prj_w[(long long)o * C_ + h * HD_ + c];
    for (int d = 0; d < HD_; ++d) {
      float s = 0.f;
#pragma unroll 8
      for (int c = 0; c < HD_; ++c) s += pw[c] * S[c][d];
      W_eff[((long long)b * C_ + o) * C_ + h * HD_ + d] = f2bf(s);
    }
  }
}

// ---------------------------------------------------------------------------
// Host launcher
// ---------------------------------------------------------------------------
extern "C" void kernel_launch(void* const* d_in, const int* in_sizes, int n_in,
                              void* d_out, int out_size, void* d_ws, size_t ws_size,
                              hipStream_t stream) {
  (void)in_sizes; (void)n_in; (void)out_size; (void)ws_size;
  const float* inputs  = (const float*)d_in[0];
  const float* d_c     = (const float*)d_in[1];
  const float* d_l     = (const float*)d_in[2];
  const float* temp    = (const float*)d_in[3];
  const float* prj_w   = (const float*)d_in[4];
  const float* prj_b   = (const float*)d_in[5];
  const float* dp_w1   = (const float*)d_in[6];
  const float* dp_b1   = (const float*)d_in[7];
  const float* dp_w2   = (const float*)d_in[8];
  const float* dp_b2   = (const float*)d_in[9];
  const float* wq_pw   = (const float*)d_in[10];
  const float* wq_pw_b = (const float*)d_in[11];
  const float* wq_dw   = (const float*)d_in[12];
  const float* wq_dw_b = (const float*)d_in[13];
  const float* wk_pw   = (const float*)d_in[14];
  const float* wk_pw_b = (const float*)d_in[15];
  const float* wk_dw   = (const float*)d_in[16];
  const float* wk_dw_b = (const float*)d_in[17];
  const float* wv_pw   = (const float*)d_in[18];
  const float* wv_pw_b = (const float*)d_in[19];
  const float* wv_dw   = (const float*)d_in[20];
  const float* wv_dw_b = (const float*)d_in[21];

  // workspace carve-up (256B aligned)
  char* ws = (char*)d_ws;
  size_t off = 0;
  auto carve = [&](size_t bytes) { size_t r = off; off = (off + bytes + 255) & ~(size_t)255; return r; };
  __bf16* Wqkv    = (__bf16*)(ws + carve((size_t)CQKV_ * C_ * 2));
  float*  biasqkv = (float*) (ws + carve((size_t)CQKV_ * 4));
  float*  norms   = (float*) (ws + carve((size_t)B_ * 2 * C_ * 4));
  float*  dcp     = (float*) (ws + carve((size_t)B_ * C_ * 4));
  __bf16* W_eff   = (__bf16*)(ws + carve((size_t)B_ * C_ * C_ * 2));
  __bf16* QKVpre  = (__bf16*)(ws + carve((size_t)B_ * CQKV_ * N_ * 2));
  __bf16* QKV     = (__bf16*)(ws + carve((size_t)B_ * CQKV_ * N_ * 2));

  // K0: weight packing + norm zeroing
  prep_kernel<<<dim3((CQKV_ * C_ + 255) / 256), 256, 0, stream>>>(
      wq_pw, wk_pw, wv_pw, wq_pw_b, wk_pw_b, wv_pw_b, Wqkv, biasqkv, norms);

  // K1: dcp MLP
  dcp_kernel<<<1, 384, 0, stream>>>(d_c, dp_w1, dp_b1, dp_w2, dp_b2, dcp);

  // K2: QKVpre = Wqkv (1152x384) x X (384x16384) per batch, bf16 out
  gemm_wmma<float, __bf16><<<dim3(N_ / 128, CQKV_ / 128, B_), 256, 0, stream>>>(
      Wqkv, 0LL, inputs, (long long)C_ * N_, biasqkv,
      QKVpre, (long long)CQKV_ * N_, C_, N_);

  // K3: depthwise 3x3 + channel sumsq for q,k
  dwconv_kernel<<<dim3(CQKV_, B_), 256, 0, stream>>>(
      QKVpre, wq_dw, wk_dw, wv_dw, wq_dw_b, wk_dw_b, wv_dw_b, QKV, norms);

  // K4: attention logits (WMMA over N) + softmax + W_eff = prj_w * blockdiag(attn)
  attn_kernel<<<dim3(HEADS_, B_), 128, 0, stream>>>(
      QKV, norms, dcp, d_l, temp, prj_w, W_eff);

  // K5: out = W_eff (384x384) x V (384x16384) + prj_b, fp32 out
  const __bf16* Vp = QKV + (long long)2 * C_ * N_;   // v block within channel dim
  gemm_wmma<__bf16, float><<<dim3(N_ / 128, C_ / 128, B_), 256, 0, stream>>>(
      W_eff, (long long)C_ * C_, Vp, (long long)CQKV_ * N_, prj_b,
      (float*)d_out, (long long)C_ * N_, C_, N_);
}
